// IterativeModelWithFirst_Con_2937757631152
// MI455X (gfx1250) — matrix operand order repaired and checked by reading, not compiled, not run on hardware
//
#include <hip/hip_runtime.h>
#include <math.h>

// ---------------------------------------------------------------------------
// Types for CDNA5 (gfx1250) WMMA: wave32, v_wmma_f32_16x16x32_bf16
// ---------------------------------------------------------------------------
typedef __bf16 bf16_t;
typedef __attribute__((ext_vector_type(16))) __bf16 v16bf;
typedef __attribute__((ext_vector_type(8)))  __bf16 v8bf;
typedef __attribute__((ext_vector_type(8)))  float  v8f;

// Problem constants (from the reference)
constexpr int Bn    = 2;
constexpr int Cc    = 1024;
constexpr int Nn    = 80;              // tokens; 80 = 5 * 16 (exact M tiles)
constexpr int Dd    = 3 * Cc + 2;      // 3074
constexpr int Dp    = 3104;            // Dd padded to multiple of 32 (= 32*97)
constexpr int DFFc  = 2048;            // already multiple of 32
constexpr int REFINEc = 3;
constexpr int CP2   = Cc + 2;          // 1026
constexpr int CP2p  = 1040;            // padded to multiple of 16
constexpr int NPAD  = 96;              // N padded to multiple of 32 (attn K dim)
constexpr int MT    = Nn / 16;         // 5 M-tiles (whole M per wave)

__device__ __forceinline__ int iclampd(int v, int lo, int hi) {
  return v < lo ? lo : (v > hi ? hi : v);
}

// ---------------------------------------------------------------------------
// bf16 WMMA GEMM:  C[m,o] = sum_k A[m,k] * Bw[o,k] (+ bias[o])
// A:  (80 x K) bf16 row-major, per-batch stride strideA.
// Bw: (Nout x K) bf16 row-major ("columns" of math-B are rows of Bw,
//     contiguous along K), per-batch stride strideB (0 for shared weights).
// Each wave owns one 16-wide N-tile and ALL 5 M-tiles: one B-fragment load
// feeds 5 independent WMMAs (5 accumulators -> no RAW chain, 5x less weight
// traffic from L2). Fragment loads of a k-step are issued before the WMMA
// group. Addresses are direct base+offset expressions (NOT pointer arrays)
// so clang keeps global address-space inference -> global_load_b128, not
// flat_load_b128 (flat would also tick DScnt and the LDS path).
// ---------------------------------------------------------------------------
template<bool BIAS, bool RELU, bool STF32, bool STBF16, bool TRANS>
__global__ __launch_bounds__(128)
void gemm_wmma(const bf16_t* __restrict__ A, long long strideA, int lda,
               const bf16_t* __restrict__ Bw, long long strideB, int ldb,
               const float* __restrict__ bias,
               float* __restrict__ Cf, long long strideCf, int ldcf,
               bf16_t* __restrict__ Cb, long long strideCb, int ldcb,
               int K, int nTilesN)
{
  const int lane  = threadIdx.x & 31;
  const int wave  = threadIdx.x >> 5;
  const int tN    = blockIdx.x * 4 + wave;
  const int batch = blockIdx.y;
  if (tN >= nTilesN) return;            // wave-uniform: EXEC stays all-ones

  const int half = lane >> 4;           // 0: lanes 0-15, 1: lanes 16-31
  const int l16  = lane & 15;

  // A fragment (16-bit A 16x32, this code's contiguous k order, not the ISA table order): row = M within tile; lanes<16
  // take K {0..15}, lanes>=16 take K {16..31}
  const bf16_t* __restrict__ Abase =
      A + (long long)batch * strideA + (long long)l16 * lda + half * 16;
  // B fragment: per-lane column (= row of Bw), 16 contiguous K values,
  // lanes<16 -> K+0..15, lanes>=16 -> K+16..31
  const bf16_t* __restrict__ Brow =
      Bw + (long long)batch * strideB + (long long)(tN * 16 + l16) * ldb
         + half * 16;

  v8f acc[MT];
  {
    const v8f z = {0.f,0.f,0.f,0.f,0.f,0.f,0.f,0.f};
#pragma unroll
    for (int m = 0; m < MT; ++m) acc[m] = z;
  }

  for (int k = 0; k < K; k += 32) {
    if (k + 128 < K) __builtin_prefetch(Brow + k + 128, 0, 3);  // global_prefetch_b8
    // Issue ALL fragment loads of this k-step (B then 5x A) before the WMMAs:
    // in-order load returns allow staged s_wait_loadcnt instead of draining.
    const v16bf bfrag = *(const v16bf*)(Brow + k);
    v16bf af[MT];
#pragma unroll
    for (int m = 0; m < MT; ++m) {
      const bf16_t* ap = Abase + (long long)(m * 16) * lda + k;
      ((v8bf*)&af[m])[0] = *(const v8bf*)(ap);
      ((v8bf*)&af[m])[1] = *(const v8bf*)(ap + 8);
    }
#pragma unroll
    for (int m = 0; m < MT; ++m)
      acc[m] = __builtin_amdgcn_wmma_f32_16x16x32_bf16(false, af[m], false, bfrag,
                                                       (short)0, acc[m], false, false);
  }

  // C/D layout (ISA 7.12.2): VGPR r -> M = r (lanes<16) or r+8 (lanes>=16),
  // N = lane & 15 for all VGPRs.
  const int col = tN * 16 + l16;
  float bv = 0.0f;
  if (BIAS) bv = bias[col];
#pragma unroll
  for (int m = 0; m < MT; ++m) {
    const int rbase = m * 16 + half * 8;
#pragma unroll
    for (int r = 0; r < 8; ++r) {
      float v = acc[m][r] + bv;
      if (RELU) v = fmaxf(v, 0.0f);
      const int row = rbase + r;
      if (STF32)
        Cf[(long long)batch * strideCf + (long long)row * ldcf + col] = v;
      if (STBF16) {
        long long o = TRANS
          ? ((long long)batch * strideCb + (long long)col * ldcb + row)
          : ((long long)batch * strideCb + (long long)row * ldcb + col);
        Cb[o] = (bf16_t)v;
      }
    }
  }
}

// ---------------------------------------------------------------------------
// fp32 -> bf16 weight conversion with zero padding (rows R->Rp, cols K0->Kp)
// ---------------------------------------------------------------------------
__global__ void convert_pad_bf16(const float* __restrict__ src, int R, int K0,
                                 bf16_t* __restrict__ dst, int Rp, int Kp)
{
  long long idx = (long long)blockIdx.x * blockDim.x + threadIdx.x;
  long long total = (long long)Rp * Kp;
  if (idx >= total) return;
  int r = (int)(idx / Kp), c = (int)(idx % Kp);
  float v = (r < R && c < K0) ? src[(long long)r * K0 + c] : 0.0f;
  dst[idx] = (bf16_t)v;
}

__global__ void pad_bias_f32(const float* __restrict__ src, int n,
                             float* __restrict__ dst, int np_)
{
  int i = blockIdx.x * blockDim.x + threadIdx.x;
  if (i < np_) dst[i] = (i < n) ? src[i] : 0.0f;
}

// ---------------------------------------------------------------------------
// Boundary roll alignment: per batch find shift minimizing L1 distance,
// then write rolled first_boundary. (argmin ties -> smallest index)
// ---------------------------------------------------------------------------
__global__ __launch_bounds__(128)
void align_first_boundary(const int* __restrict__ prev_b,
                          const int* __restrict__ first_b,
                          int* __restrict__ first_roll)
{
  __shared__ int sd[128];
  __shared__ int si[128];
  const int b = blockIdx.x;
  const int t = threadIdx.x;
  const int* pb = prev_b  + b * Nn * 2;
  const int* fb = first_b + b * Nn * 2;
  int bestd = 0x7fffffff, besti = 0;
  for (int i = t; i < Nn; i += 128) {
    int d = 0;
    for (int j = 0; j < Nn; ++j) {
      int jj = (j - i + Nn) % Nn;
      int d0 = pb[j*2+0] - fb[jj*2+0]; if (d0 < 0) d0 = -d0;
      int d1 = pb[j*2+1] - fb[jj*2+1]; if (d1 < 0) d1 = -d1;
      d += d0 + d1;
    }
    if (d < bestd) { bestd = d; besti = i; }
  }
  sd[t] = bestd; si[t] = besti;
  __syncthreads();
  for (int s = 64; s > 0; s >>= 1) {
    if (t < s) {
      if (sd[t+s] < sd[t] || (sd[t+s] == sd[t] && si[t+s] < si[t])) {
        sd[t] = sd[t+s]; si[t] = si[t+s];
      }
    }
    __syncthreads();
  }
  const int best = si[0];
  for (int j = t; j < Nn; j += 128) {
    int jj = (j - best + Nn) % Nn;
    first_roll[(b*Nn + j)*2 + 0] = fb[jj*2 + 0];
    first_roll[(b*Nn + j)*2 + 1] = fb[jj*2 + 1];
  }
}

// ---------------------------------------------------------------------------
// Bilinear sample of a 14x14 map at an integer pixel of the 224x224 upsample
// (half-pixel centers, edge clamp == jax.image.resize 'bilinear' upscaling)
// ---------------------------------------------------------------------------
__device__ __forceinline__ float bilin14(const float* __restrict__ fp,
                                         int y, int x)
{
  float fy = ((float)y + 0.5f) * (1.0f / 16.0f) - 0.5f;
  float fx = ((float)x + 0.5f) * (1.0f / 16.0f) - 0.5f;
  float y0f = floorf(fy), x0f = floorf(fx);
  float wy = fy - y0f, wx = fx - x0f;
  int y0 = (int)y0f, x0 = (int)x0f;
  int y0c = iclampd(y0, 0, 13),     y1c = iclampd(y0 + 1, 0, 13);
  int x0c = iclampd(x0, 0, 13),     x1c = iclampd(x0 + 1, 0, 13);
  float v00 = fp[y0c*14 + x0c], v01 = fp[y0c*14 + x1c];
  float v10 = fp[y1c*14 + x0c], v11 = fp[y1c*14 + x1c];
  return (1.f - wy) * ((1.f - wx) * v00 + wx * v01)
       +        wy  * ((1.f - wx) * v10 + wx * v11);
}

__global__ void gather_feats_i(const float* __restrict__ feat,   // (B,C,14,14)
                               const int* __restrict__ bnd,      // (B,N,2)
                               float* __restrict__ out)          // (B,N,C)
{
  int b = blockIdx.z, n = blockIdx.y;
  int c = blockIdx.x * blockDim.x + threadIdx.x;
  if (c >= Cc) return;
  int y = bnd[(b*Nn + n)*2 + 0];
  int x = bnd[(b*Nn + n)*2 + 1];
  const float* fp = feat + ((long long)b*Cc + c) * 196;
  out[((long long)b*Nn + n)*Cc + c] = bilin14(fp, y, x);
}

__global__ void gather_feats_f(const float* __restrict__ feat,
                               const float* __restrict__ pos,    // (B,N,2) float
                               float* __restrict__ out)
{
  int b = blockIdx.z, n = blockIdx.y;
  int c = blockIdx.x * blockDim.x + threadIdx.x;
  if (c >= Cc) return;
  int y = (int)pos[(b*Nn + n)*2 + 0];
  int x = (int)pos[(b*Nn + n)*2 + 1];
  const float* fp = feat + ((long long)b*Cc + c) * 196;
  out[((long long)b*Nn + n)*Cc + c] = bilin14(fp, y, x);
}

__global__ void init_curr(const int* __restrict__ prev_b, float* __restrict__ curr)
{
  int i = blockIdx.x * blockDim.x + threadIdx.x;
  if (i < Bn * Nn * 2) curr[i] = (float)prev_b[i];
}

// ---------------------------------------------------------------------------
// Token assembly: concat [pre_q | bf | curr | first_q] -> LayerNorm -> +PE,
// emitting fp32 (for residual) and zero-padded bf16 (for WMMA).
// ---------------------------------------------------------------------------
__global__ __launch_bounds__(256)
void build_tokens(const float* __restrict__ pre_q, const float* __restrict__ bfq,
                  const float* __restrict__ curr, const float* __restrict__ first_q,
                  const float* __restrict__ g,    const float* __restrict__ bb,
                  float* __restrict__ tok_f, bf16_t* __restrict__ tok_b)
{
  __shared__ float ssum[256], ssq[256];
  const int bn = blockIdx.x;            // b*Nn + n
  const int n  = bn % Nn;
  const int t  = threadIdx.x;

  auto elem = [&](int d) -> float {
    if (d < Cc)        return pre_q[(long long)bn*Cc + d];
    if (d < 2*Cc)      return bfq[(long long)bn*Cc + (d - Cc)];
    if (d < 2*Cc + 2)  return curr[bn*2 + (d - 2*Cc)];
    return first_q[(long long)bn*Cc + (d - 2*Cc - 2)];
  };

  float s = 0.f, sq = 0.f;
  for (int d = t; d < Dd; d += 256) { float v = elem(d); s += v; sq += v * v; }
  ssum[t] = s; ssq[t] = sq;
  __syncthreads();
  for (int st = 128; st > 0; st >>= 1) {
    if (t < st) { ssum[t] += ssum[t+st]; ssq[t] += ssq[t+st]; }
    __syncthreads();
  }
  const float mean = ssum[0] / (float)Dd;
  const float var  = ssq[0] / (float)Dd - mean * mean;
  const float inv  = rsqrtf(var + 1e-5f);

  const float k_pe = -logf(10000.0f) / (float)Dd;
  for (int d = t; d < Dp; d += 256) {
    float o = 0.0f;
    if (d < Dd) {
      float v  = elem(d);
      float ln = g[d] * (v - mean) * inv + bb[d];
      int   i2 = (d >> 1) * 2;
      float ang = (float)n * expf(k_pe * (float)i2);
      float pe  = (d & 1) ? cosf(ang) : sinf(ang);
      o = ln + pe;
    }
    tok_f[(long long)bn*Dp + d] = o;
    tok_b[(long long)bn*Dp + d] = (bf16_t)o;
  }
}

// residual add + LayerNorm -> fp32 + zero-padded bf16
__global__ __launch_bounds__(256)
void add_ln(const float* __restrict__ base, const float* __restrict__ add,
            const float* __restrict__ g, const float* __restrict__ bb,
            float* __restrict__ out_f, bf16_t* __restrict__ out_b)
{
  __shared__ float ssum[256], ssq[256];
  const int bn = blockIdx.x;
  const int t  = threadIdx.x;
  const float* pb = base + (long long)bn * Dp;
  const float* pa = add  + (long long)bn * Dp;

  float s = 0.f, sq = 0.f;
  for (int d = t; d < Dd; d += 256) { float v = pb[d] + pa[d]; s += v; sq += v*v; }
  ssum[t] = s; ssq[t] = sq;
  __syncthreads();
  for (int st = 128; st > 0; st >>= 1) {
    if (t < st) { ssum[t] += ssum[t+st]; ssq[t] += ssq[t+st]; }
    __syncthreads();
  }
  const float mean = ssum[0] / (float)Dd;
  const float var  = ssq[0] / (float)Dd - mean * mean;
  const float inv  = rsqrtf(var + 1e-5f);

  for (int d = t; d < Dp; d += 256) {
    float o = 0.0f;
    if (d < Dd) {
      float v = pb[d] + pa[d];
      o = g[d] * (v - mean) * inv + bb[d];
    }
    out_f[(long long)bn*Dp + d] = o;
    out_b[(long long)bn*Dp + d] = (bf16_t)o;
  }
}

// softmax over 80 cols (scaled) -> bf16 with zero pad to NPAD
__global__ __launch_bounds__(128)
void softmax_rows(const float* __restrict__ scores, bf16_t* __restrict__ attn)
{
  __shared__ float sh[128];
  const int bn = blockIdx.x;
  const int t  = threadIdx.x;
  const float scale = rsqrtf((float)Dd);
  const float* row = scores + (long long)bn * Nn;

  float m = -1e30f;
  for (int j = t; j < Nn; j += 128) m = fmaxf(m, row[j] * scale);
  sh[t] = m; __syncthreads();
  for (int s = 64; s > 0; s >>= 1) { if (t < s) sh[t] = fmaxf(sh[t], sh[t+s]); __syncthreads(); }
  m = sh[0]; __syncthreads();

  float sum = 0.f;
  for (int j = t; j < Nn; j += 128) sum += expf(row[j] * scale - m);
  sh[t] = sum; __syncthreads();
  for (int s = 64; s > 0; s >>= 1) { if (t < s) sh[t] += sh[t+s]; __syncthreads(); }
  const float invs = 1.0f / sh[0];

  for (int j = t; j < NPAD; j += 128) {
    float v = (j < Nn) ? expf(row[j] * scale - m) * invs : 0.0f;
    attn[(long long)bn * NPAD + j] = (bf16_t)v;
  }
}

// curr += out[:, C:], clip to [0,223]; pre_q += out[:, :C]; emit results[r]
__global__ __launch_bounds__(256)
void update_state(const float* __restrict__ outbuf,   // (B,N,CP2p)
                  float* __restrict__ curr, float* __restrict__ pre_q,
                  float* __restrict__ results, int r)
{
  const int bn = blockIdx.x;
  const int t  = threadIdx.x;
  const float* row = outbuf + (long long)bn * CP2p;
  for (int c = t; c < Cc; c += 256)
    pre_q[(long long)bn*Cc + c] += row[c];
  if (t < 2) {
    float v = curr[bn*2 + t] + row[Cc + t];
    v = fminf(fmaxf(v, 0.0f), 223.0f);
    curr[bn*2 + t] = v;
    results[((long long)r * Bn * Nn + bn) * 2 + t] = v;
  }
}

// ---------------------------------------------------------------------------
// Host-side orchestration (all on `stream`; capture-safe)
// ---------------------------------------------------------------------------
extern "C" void kernel_launch(void* const* d_in, const int* in_sizes, int n_in,
                              void* d_out, int out_size, void* d_ws, size_t ws_size,
                              hipStream_t stream)
{
  (void)in_sizes; (void)n_in; (void)out_size; (void)ws_size;

  const float* pre_img   = (const float*)d_in[0];
  const float* curr_img  = (const float*)d_in[1];
  const float* first_img = (const float*)d_in[2];
  const int*   prev_b    = (const int*)  d_in[3];
  const int*   first_b   = (const int*)  d_in[4];
  const float* ln_g = (const float*)d_in[5];
  const float* ln_b = (const float*)d_in[6];
  const float* ipw  = (const float*)d_in[7];
  const float* ipb  = (const float*)d_in[8];
  const float* opw  = (const float*)d_in[9];
  const float* opb  = (const float*)d_in[10];
  const float* n1g  = (const float*)d_in[11];
  const float* n1b  = (const float*)d_in[12];
  const float* l1w  = (const float*)d_in[13];
  const float* l1b  = (const float*)d_in[14];
  const float* l2w  = (const float*)d_in[15];
  const float* l2b  = (const float*)d_in[16];
  const float* n2g  = (const float*)d_in[17];
  const float* n2b  = (const float*)d_in[18];
  const float* ow   = (const float*)d_in[19];
  const float* ob   = (const float*)d_in[20];
  float* results = (float*)d_out;

  // ---- workspace carve-out (256B aligned) ----
  char* wp = (char*)d_ws;
  auto alloc = [&](size_t bytes) -> char* {
    char* q = wp; wp += (bytes + 255) & ~(size_t)255; return q;
  };
  bf16_t* Wq  = (bf16_t*)alloc((size_t)Dp   * Dp   * sizeof(bf16_t));
  bf16_t* Wk  = (bf16_t*)alloc((size_t)Dp   * Dp   * sizeof(bf16_t));
  bf16_t* Wv  = (bf16_t*)alloc((size_t)Dp   * Dp   * sizeof(bf16_t));
  bf16_t* Wop = (bf16_t*)alloc((size_t)Dp   * Dp   * sizeof(bf16_t));
  bf16_t* Wl1 = (bf16_t*)alloc((size_t)DFFc * Dp   * sizeof(bf16_t));
  bf16_t* Wl2 = (bf16_t*)alloc((size_t)Dp   * DFFc * sizeof(bf16_t));
  bf16_t* Wow = (bf16_t*)alloc((size_t)CP2p * Dp   * sizeof(bf16_t));
  float* bq  = (float*)alloc(Dp   * sizeof(float));
  float* bk  = (float*)alloc(Dp   * sizeof(float));
  float* bv_ = (float*)alloc(Dp   * sizeof(float));
  float* bop = (float*)alloc(Dp   * sizeof(float));
  float* bl1 = (float*)alloc(DFFc * sizeof(float));
  float* bl2 = (float*)alloc(Dp   * sizeof(float));
  float* bow = (float*)alloc(CP2p * sizeof(float));
  int*   first_roll = (int*)alloc((size_t)Bn*Nn*2 * sizeof(int));
  float* pre_q   = (float*)alloc((size_t)Bn*Nn*Cc * sizeof(float));
  float* first_q = (float*)alloc((size_t)Bn*Nn*Cc * sizeof(float));
  float* bfq     = (float*)alloc((size_t)Bn*Nn*Cc * sizeof(float));
  float* curr    = (float*)alloc((size_t)Bn*Nn*2  * sizeof(float));
  float*  tok_f = (float*) alloc((size_t)Bn*Nn*Dp * sizeof(float));
  bf16_t* tok_b = (bf16_t*)alloc((size_t)Bn*Nn*Dp * sizeof(bf16_t));
  bf16_t* qb    = (bf16_t*)alloc((size_t)Bn*Nn*Dp * sizeof(bf16_t));
  bf16_t* kb    = (bf16_t*)alloc((size_t)Bn*Nn*Dp * sizeof(bf16_t));
  bf16_t* vT    = (bf16_t*)alloc((size_t)Bn*Dp*NPAD * sizeof(bf16_t));
  float*  scores= (float*) alloc((size_t)Bn*Nn*Nn * sizeof(float));
  bf16_t* attn  = (bf16_t*)alloc((size_t)Bn*Nn*NPAD * sizeof(bf16_t));
  bf16_t* abuf  = (bf16_t*)alloc((size_t)Bn*Nn*Dp * sizeof(bf16_t));
  float*  proj  = (float*) alloc((size_t)Bn*Nn*Dp * sizeof(float));
  float*  xf    = (float*) alloc((size_t)Bn*Nn*Dp * sizeof(float));
  bf16_t* xb    = (bf16_t*)alloc((size_t)Bn*Nn*Dp * sizeof(bf16_t));
  bf16_t* ffb   = (bf16_t*)alloc((size_t)Bn*Nn*DFFc * sizeof(bf16_t));
  float*  ff2   = (float*) alloc((size_t)Bn*Nn*Dp * sizeof(float));
  float*  x2f   = (float*) alloc((size_t)Bn*Nn*Dp * sizeof(float));
  bf16_t* x2b   = (bf16_t*)alloc((size_t)Bn*Nn*Dp * sizeof(bf16_t));
  float*  outbuf= (float*) alloc((size_t)Bn*Nn*CP2p * sizeof(float));

  // ---- once-per-call weight conversion to padded bf16 (stays hot in L2) ----
  {
    long long totD = (long long)Dp * Dp;
    int gD = (int)((totD + 255) / 256);
    convert_pad_bf16<<<gD,256,0,stream>>>(ipw,                    Dd, Dd, Wq,  Dp, Dp);
    convert_pad_bf16<<<gD,256,0,stream>>>(ipw + (long long)Dd*Dd, Dd, Dd, Wk,  Dp, Dp);
    convert_pad_bf16<<<gD,256,0,stream>>>(ipw + 2ll*Dd*Dd,        Dd, Dd, Wv,  Dp, Dp);
    convert_pad_bf16<<<gD,256,0,stream>>>(opw,                    Dd, Dd, Wop, Dp, Dp);
    long long t1 = (long long)DFFc * Dp;
    convert_pad_bf16<<<(int)((t1+255)/256),256,0,stream>>>(l1w, DFFc, Dd, Wl1, DFFc, Dp);
    long long t2 = (long long)Dp * DFFc;
    convert_pad_bf16<<<(int)((t2+255)/256),256,0,stream>>>(l2w, Dd, DFFc, Wl2, Dp, DFFc);
    long long t3 = (long long)CP2p * Dp;
    convert_pad_bf16<<<(int)((t3+255)/256),256,0,stream>>>(ow, CP2, Dd, Wow, CP2p, Dp);
    pad_bias_f32<<<(Dp+255)/256,256,0,stream>>>(ipb,            Dd,   bq,  Dp);
    pad_bias_f32<<<(Dp+255)/256,256,0,stream>>>(ipb + Dd,       Dd,   bk,  Dp);
    pad_bias_f32<<<(Dp+255)/256,256,0,stream>>>(ipb + 2*Dd,     Dd,   bv_, Dp);
    pad_bias_f32<<<(Dp+255)/256,256,0,stream>>>(opb,            Dd,   bop, Dp);
    pad_bias_f32<<<(DFFc+255)/256,256,0,stream>>>(l1b,          DFFc, bl1, DFFc);
    pad_bias_f32<<<(Dp+255)/256,256,0,stream>>>(l2b,            Dd,   bl2, Dp);
    pad_bias_f32<<<(CP2p+255)/256,256,0,stream>>>(ob,           CP2,  bow, CP2p);
  }

  // ---- setup: roll alignment, static gathers, curr init, vT pad clear ----
  align_first_boundary<<<Bn,128,0,stream>>>(prev_b, first_b, first_roll);
  gather_feats_i<<<dim3(Cc/256,Nn,Bn),256,0,stream>>>(pre_img,   prev_b,     pre_q);
  gather_feats_i<<<dim3(Cc/256,Nn,Bn),256,0,stream>>>(first_img, first_roll, first_q);
  init_curr<<<(Bn*Nn*2+255)/256,256,0,stream>>>(prev_b, curr);
  hipMemsetAsync(vT, 0, (size_t)Bn*Dp*NPAD*sizeof(bf16_t), stream);

  const long long sTok = (long long)Nn * Dp;     // per-batch activation stride
  const long long sVT  = (long long)Dp * NPAD;
  const int nTD  = Dp  / 16;    // 194
  const int nTN  = Nn  / 16;    // 5
  const int nTF  = DFFc/ 16;    // 128
  const int nTO  = CP2p/ 16;    // 65

  for (int r = 0; r < REFINEc; ++r) {
    gather_feats_f<<<dim3(Cc/256,Nn,Bn),256,0,stream>>>(curr_img, curr, bfq);
    build_tokens<<<Bn*Nn,256,0,stream>>>(pre_q, bfq, curr, first_q, ln_g, ln_b,
                                         tok_f, tok_b);
    // Q = tok @ Wq^T + bq  (bf16 out)
    gemm_wmma<true,false,false,true,false><<<dim3((nTD+3)/4,Bn),128,0,stream>>>(
        tok_b, sTok, Dp, Wq, 0, Dp, bq, nullptr,0,0, qb, sTok, Dp, Dp, nTD);
    // K
    gemm_wmma<true,false,false,true,false><<<dim3((nTD+3)/4,Bn),128,0,stream>>>(
        tok_b, sTok, Dp, Wk, 0, Dp, bk, nullptr,0,0, kb, sTok, Dp, Dp, nTD);
    // V (stored transposed: vT[o, token])
    gemm_wmma<true,false,false,true,true><<<dim3((nTD+3)/4,Bn),128,0,stream>>>(
        tok_b, sTok, Dp, Wv, 0, Dp, bv_, nullptr,0,0, vT, sVT, NPAD, Dp, nTD);
    // scores = Q @ K^T  (fp32)
    gemm_wmma<false,false,true,false,false><<<dim3((nTN+3)/4,Bn),128,0,stream>>>(
        qb, sTok, Dp, kb, sTok, Dp, nullptr,
        scores, (long long)Nn*Nn, Nn, nullptr,0,0, Dp, nTN);
    softmax_rows<<<Bn*Nn,128,0,stream>>>(scores, attn);
    // A = attn @ V  (bf16 out)
    gemm_wmma<false,false,false,true,false><<<dim3((nTD+3)/4,Bn),128,0,stream>>>(
        attn, (long long)Nn*NPAD, NPAD, vT, sVT, NPAD, nullptr,
        nullptr,0,0, abuf, sTok, Dp, NPAD, nTD);
    // out-proj (fp32, + bias)
    gemm_wmma<true,false,true,false,false><<<dim3((nTD+3)/4,Bn),128,0,stream>>>(
        abuf, sTok, Dp, Wop, 0, Dp, bop, proj, sTok, Dp, nullptr,0,0, Dp, nTD);
    add_ln<<<Bn*Nn,256,0,stream>>>(tok_f, proj, n1g, n1b, xf, xb);
    // FFN up + ReLU (bf16 out)
    gemm_wmma<true,true,false,true,false><<<dim3((nTF+3)/4,Bn),128,0,stream>>>(
        xb, sTok, Dp, Wl1, 0, Dp, bl1, nullptr,0,0,
        ffb, (long long)Nn*DFFc, DFFc, Dp, nTF);
    // FFN down (fp32)
    gemm_wmma<true,false,true,false,false><<<dim3((nTD+3)/4,Bn),128,0,stream>>>(
        ffb, (long long)Nn*DFFc, DFFc, Wl2, 0, DFFc, bl2,
        ff2, sTok, Dp, nullptr,0,0, DFFc, nTD);
    add_ln<<<Bn*Nn,256,0,stream>>>(xf, ff2, n2g, n2b, x2f, x2b);
    // output head (fp32)
    gemm_wmma<true,false,true,false,false><<<dim3((nTO+3)/4,Bn),128,0,stream>>>(
        x2b, sTok, Dp, Wow, 0, Dp, bow,
        outbuf, (long long)Nn*CP2p, CP2p, nullptr,0,0, Dp, nTO);
    update_state<<<Bn*Nn,256,0,stream>>>(outbuf, curr, pre_q, results, r);
  }
}
